// FrequencyAttention_55216099557499
// MI455X (gfx1250) — matrix operand order repaired and checked by reading, not compiled, not run on hardware
//
#include <hip/hip_runtime.h>

typedef _Float16 v16h __attribute__((ext_vector_type(16)));
typedef _Float16 v8h  __attribute__((ext_vector_type(8)));
typedef float    v8f  __attribute__((ext_vector_type(8)));
typedef unsigned int v4u __attribute__((ext_vector_type(4)));
typedef int      v4i __attribute__((ext_vector_type(4)));
typedef int      v8i __attribute__((ext_vector_type(8)));

namespace {
constexpr int NB    = 32;          // batch
constexpr int NL    = 4096;        // sequence length
constexpr int ND    = 256;         // d_model
constexpr int NMODE = 64;          // kept rfft modes
constexpr int MF    = 2 * NMODE;   // 128 stacked Re|Im rows
constexpr float PI2 = 6.283185307179586f;
}

// ---------------------------------------------------------------------------
// Twiddle tables (f16).
//   Tf[fs][l], fs<64 :  cos(2*pi*f*l/N)        (Re rows of forward DFT)
//             fs>=64 : -sin(2*pi*f*l/N)        (Im rows)
//   Ti[l][fs], fs<64 :  cos(2*pi*f*l/N)*c_f    (inverse, c_0=0.5 else 1)
//             fs>=64 : -sin(2*pi*f*l/N)*c_f    (f=0 sin column is 0 ->
//                                               irfft discards Im(X0) for free)
// Phase reduced exactly with integer (f*l) mod N before sincos.
// ---------------------------------------------------------------------------
__global__ void __launch_bounds__(256)
k_twiddles(_Float16* __restrict__ Tf, _Float16* __restrict__ Ti)
{
    const int idx = blockIdx.x * blockDim.x + threadIdx.x;   // MF*NL threads
    const int l   = idx & (NL - 1);
    const int fs  = idx >> 12;
    const int f   = fs & (NMODE - 1);
    const int ph  = (f * l) & (NL - 1);
    const float th = (float)ph * (PI2 / (float)NL);
    float s, c;
    __sincosf(th, &s, &c);
    const float v = (fs < NMODE) ? c : -s;
    Tf[(size_t)fs * NL + l] = (_Float16)v;
    Ti[(size_t)l * MF + fs] = (_Float16)((f == 0) ? 0.5f * v : v);
}

// ---------------------------------------------------------------------------
// Stage 1: forward DFT.  X1[b](128x256) = Tf(128x4096) * f16(x[b])(4096x256)
// Workgroup: 256 threads = 8 waves covering [128 M x 32 N]; x tile is
// converted f32->f16 while being staged to LDS transposed (pitch 40 halfs =
// 80B -> 16B-aligned b128 fragment reads, conflict-controlled).
// ---------------------------------------------------------------------------
template<int BN>
__global__ void __launch_bounds__(256)
k_dft(const _Float16* __restrict__ A, const float* __restrict__ Bsrc,
      _Float16* __restrict__ Cdst, int K, int N)
{
    constexpr int ACC   = BN / 16;
    constexpr int BK    = 32;
    constexpr int PITCH = 40;                      // halfs per LDS row (80B)
    __shared__ alignas(16) _Float16 lds[BN * PITCH];

    const int nblocks = N / BN;
    const int b     = blockIdx.x / nblocks;
    const int nbase = (blockIdx.x % nblocks) * BN;

    const int t    = threadIdx.x;
    const int lane = t & 31;
    const int wave = t >> 5;
    const int hw   = lane >> 4;                    // half-wave select
    const int ln15 = lane & 15;

    const _Float16* arow = A + (size_t)(wave * 16 + ln15) * K;
    const float*    Bf   = Bsrc + (size_t)b * K * N;

    v8f acc[ACC];
    #pragma unroll
    for (int j = 0; j < ACC; ++j)
        #pragma unroll
        for (int i = 0; i < 8; ++i) acc[j][i] = 0.0f;

    for (int kb = 0; kb < K; kb += BK) {
        // Stage B tile [BK x BN] into LDS transposed: lds[n*PITCH + k]
        #pragma unroll
        for (int e = t; e < BK * BN; e += 256) {
            const int k = e / BN, n = e % BN;
            lds[n * PITCH + k] =
                (_Float16)Bf[(size_t)(kb + k) * N + (nbase + n)];
        }
        __syncthreads();

        // A fragment: lane ln15 = M row; halves map K = h + 16*hw
        const v8h alo = *reinterpret_cast<const v8h*>(arow + kb + hw * 16);
        const v8h ahi = *reinterpret_cast<const v8h*>(arow + kb + hw * 16 + 8);
        const v16h a  = __builtin_shufflevector(alo, ahi,
                          0,1,2,3,4,5,6,7,8,9,10,11,12,13,14,15);

        #pragma unroll
        for (int j = 0; j < ACC; ++j) {
            // B fragment: lane ln15 = N col; halves map K = h + 16*hw
            const _Float16* col = &lds[(j * 16 + ln15) * PITCH + hw * 16];
            const v8h blo = *reinterpret_cast<const v8h*>(col);
            const v8h bhi = *reinterpret_cast<const v8h*>(col + 8);
            const v16h bb = __builtin_shufflevector(blo, bhi,
                              0,1,2,3,4,5,6,7,8,9,10,11,12,13,14,15);
            acc[j] = __builtin_amdgcn_wmma_f32_16x16x32_f16(
                         false, a, false, bb, (short)0, acc[j], false, false);
        }
        __syncthreads();
    }

    // D fragment: M = v + 8*hw, N = ln15
    const int mrow = wave * 16 + hw * 8;
    const int ncol = nbase + ln15;
    #pragma unroll
    for (int j = 0; j < ACC; ++j)
        #pragma unroll
        for (int v = 0; v < 8; ++v)
            Cdst[((size_t)b * MF + (mrow + v)) * N + (ncol + j * 16)] =
                (_Float16)(acc[j][v]);
}

// ---------------------------------------------------------------------------
// Stage 2: per-mode complex mixing as one real GEMM per mode:
//   [XOr|XOi](32x512) = [Xr|Xi](32x512) * [[Wr,Wi],[-Wi,Wr]](512x512)
// Workgroup: 128 threads = 4 waves, covers [32 M(=batch) x 32 N]; grid =
// 64 modes x 16 N-blocks. A gathered from X1 (f16), B gathered from
// w_real/w_imag f32 with quadrant sign, converted to f16 on the fly.
// ---------------------------------------------------------------------------
__device__ inline v8h ldx1(const _Float16* __restrict__ X1, int b, int mode, int i0)
{
    const int fs = (i0 < ND) ? mode : NMODE + mode;   // Re rows then Im rows
    const int d0 = i0 & (ND - 1);
    return *reinterpret_cast<const v8h*>(X1 + ((size_t)b * MF + fs) * ND + d0);
}

__global__ void __launch_bounds__(128)
k_mix(const _Float16* __restrict__ X1, const float* __restrict__ wr,
      const float* __restrict__ wi, _Float16* __restrict__ X2)
{
    const int wg    = blockIdx.x;
    const int mode  = wg >> 4;
    const int nbase = (wg & 15) * 32;
    const int t     = threadIdx.x;
    const int lane  = t & 31;
    const int wave  = t >> 5;
    const int hw    = lane >> 4;
    const int ln15  = lane & 15;
    const int mt    = wave & 1;     // 16-row batch tile
    const int nt    = wave >> 1;    // 16-col tile within the 32-wide block

    const int  j    = nbase + nt * 16 + ln15;       // column in [0,512)
    const bool jlow = j < ND;                       // Re vs Im output half
    const int  jj   = jlow ? j : j - ND;
    const int  fsj  = jlow ? mode : NMODE + mode;

    v8f acc;
    #pragma unroll
    for (int i = 0; i < 8; ++i) acc[i] = 0.0f;

    const int brow = mt * 16 + ln15;                // M row = batch index

    for (int kb = 0; kb < 2 * ND; kb += 32) {
        const int k0 = kb + hw * 16;
        const v8h alo = ldx1(X1, brow, mode, k0);
        const v8h ahi = ldx1(X1, brow, mode, k0 + 8);
        const v16h a  = __builtin_shufflevector(alo, ahi,
                          0,1,2,3,4,5,6,7,8,9,10,11,12,13,14,15);

        const bool ilow = kb < ND;
        const int  i0   = kb + hw * 16 - (ilow ? 0 : ND);
        const float* wsrc; float sgn = 1.0f;
        if (ilow)       wsrc = jlow ? wr : wi;           //  Wr |  Wi
        else if (jlow) { wsrc = wi; sgn = -1.0f; }       // -Wi
        else            wsrc = wr;                       //  Wr
        const float* p = wsrc + ((size_t)mode * ND + i0) * ND + jj;

        v16h bb;
        #pragma unroll
        for (int h = 0; h < 16; ++h) bb[h] = (_Float16)(sgn * p[(size_t)h * ND]);

        acc = __builtin_amdgcn_wmma_f32_16x16x32_f16(
                  false, a, false, bb, (short)0, acc, false, false);
    }

    const int b0 = mt * 16 + hw * 8;
    #pragma unroll
    for (int v = 0; v < 8; ++v)
        X2[((size_t)(b0 + v) * MF + fsj) * ND + jj] = (_Float16)(acc[v]);
}

// ---------------------------------------------------------------------------
// Stage 3: inverse DFT via the Tensor Data Mover.
//   out[b](4096x256) = (2/N) * Ti(4096x128) * X2[b](128x256)
// Wave 0 issues a 2D tensor_load_to_lds per K-tile (double buffered):
//   tile 64x32 f16 from tensor [128][256], stride 256, padded in LDS with
//   1 DWORD every 32 DWORDs -> pitch 132B (bank-conflict break).
// ---------------------------------------------------------------------------
__device__ inline void tdm_load_tile(unsigned lds_off, const _Float16* gsrc)
{
    const unsigned long long ga = (unsigned long long)(uintptr_t)gsrc;
    v4u g0;
    g0[0] = 1u;                                   // count=1, user mode
    g0[1] = lds_off;                              // LDS byte address
    g0[2] = (unsigned)ga;                         // global_addr[31:0]
    g0[3] = (unsigned)((ga >> 32) & 0x01FFFFFFu)  // global_addr[56:32]
          | (2u << 30);                           // type = 2 (image)
    v8i g1;
    g1[0] = (int)((1u << 16)                      // data_size = 2 bytes
                | (1u << 20)                      // pad_enable
                | (4u << 22)                      // pad_interval = 32 DWORDs
                | (0u << 25));                    // pad_amount   = 1 DWORD
    g1[1] = (int)(256u << 16);                    // tensor_dim0 = 256
    g1[2] = (int)(128u << 16);                    // tensor_dim1 = 128
    g1[3] = (int)(64u << 16);                     // tile_dim0 = 64
    g1[4] = 32;                                   // tile_dim1 = 32, tile_dim2=0
    g1[5] = 256;                                  // tensor_dim0_stride = 256
    g1[6] = 0;
    g1[7] = 0;
    const v4i gz4 = {0, 0, 0, 0};                 // groups 2/3: 2D tensor
    const v8i gz8 = {0, 0, 0, 0, 0, 0, 0, 0};
    __builtin_amdgcn_tensor_load_to_lds(g0, g1, gz4, gz4, gz8, 0);
}

__global__ void __launch_bounds__(256)
k_idft(const _Float16* __restrict__ A,      // Ti [4096][128]
       const _Float16* __restrict__ X2,     // [32][128][256]
       float* __restrict__ out, float scale)
{
    constexpr int K     = MF;                // 128
    constexpr int BK    = 32;
    constexpr int BN    = 64;
    constexpr int ACC   = BN / 16;
    constexpr int NK    = K / BK;            // 4
    constexpr int PITCH = 66;                // halfs per LDS row (132B)
    __shared__ alignas(16) _Float16 tile[2][BK * PITCH];

    const int wg    = blockIdx.x;
    const int b     = wg >> 7;               // 32 mblk * 4 nblk = 128 per batch
    const int rem   = wg & 127;
    const int mbase = (rem >> 2) * 128;
    const int nbase = (rem & 3) * BN;

    const int t    = threadIdx.x;
    const int lane = t & 31;
    const int wave = t >> 5;
    const int hw   = lane >> 4;
    const int ln15 = lane & 15;

    const _Float16* arow = A + (size_t)(mbase + wave * 16 + ln15) * K;
    const _Float16* bsrc = X2 + (size_t)b * K * ND + nbase;   // tile origin, row 0

    const unsigned lds0 = (unsigned)(uintptr_t)&tile[0][0];   // LDS byte offset
    const unsigned lds1 = (unsigned)(uintptr_t)&tile[1][0];

    if (wave == 0) tdm_load_tile(lds0, bsrc);                 // prologue

    v8f acc[ACC];
    #pragma unroll
    for (int j = 0; j < ACC; ++j)
        #pragma unroll
        for (int i = 0; i < 8; ++i) acc[j][i] = 0.0f;

    #pragma unroll
    for (int it = 0; it < NK; ++it) {
        const int kb = it * BK;
        if (wave == 0) {
            if (it + 1 < NK) {
                // issue next tile into the other buffer, then require the
                // current tile (issued earlier, in-order) to be complete
                tdm_load_tile((it & 1) ? lds0 : lds1,
                              bsrc + (size_t)(kb + BK) * ND);
                __builtin_amdgcn_s_wait_tensorcnt(1);
            } else {
                __builtin_amdgcn_s_wait_tensorcnt(0);
            }
        }
        __syncthreads();                     // publish tile[it&1] to all waves

        const _Float16* lb = tile[it & 1];

        const v8h alo = *reinterpret_cast<const v8h*>(arow + kb + hw * 16);
        const v8h ahi = *reinterpret_cast<const v8h*>(arow + kb + hw * 16 + 8);
        const v16h a  = __builtin_shufflevector(alo, ahi,
                          0,1,2,3,4,5,6,7,8,9,10,11,12,13,14,15);

        #pragma unroll
        for (int j = 0; j < ACC; ++j) {
            // B fragment from row-major padded tile: K = h + 16*hw, N = ln15
            v16h bb;
            #pragma unroll
            for (int h = 0; h < 16; ++h)
                bb[h] = lb[(hw * 16 + h) * PITCH + (j * 16 + ln15)];
            acc[j] = __builtin_amdgcn_wmma_f32_16x16x32_f16(
                         false, a, false, bb, (short)0, acc[j], false, false);
        }
        __syncthreads();                     // tile[it&1] free for reuse
    }

    const int mrow = mbase + wave * 16 + hw * 8;
    const int ncol = nbase + ln15;
    #pragma unroll
    for (int j = 0; j < ACC; ++j)
        #pragma unroll
        for (int v = 0; v < 8; ++v)
            out[((size_t)b * NL + (mrow + v)) * ND + (ncol + j * 16)] =
                acc[j][v] * scale;
}

// ---------------------------------------------------------------------------
// Workspace (6 MB):
//   Tf [128][4096] f16 @ 0        (1 MB)
//   Ti [4096][128] f16 @ 1 MB     (1 MB)
//   X1 [32][128][256] f16 @ 2 MB  (2 MB)   forward-DFT output (Re|Im rows)
//   X2 [32][128][256] f16 @ 4 MB  (2 MB)   mixed spectrum      (Re|Im rows)
// ---------------------------------------------------------------------------
extern "C" void kernel_launch(void* const* d_in, const int* in_sizes, int n_in,
                              void* d_out, int out_size, void* d_ws, size_t ws_size,
                              hipStream_t stream)
{
    (void)in_sizes; (void)n_in; (void)out_size; (void)ws_size;
    const float* x  = (const float*)d_in[0];
    const float* wr = (const float*)d_in[1];
    const float* wi = (const float*)d_in[2];
    float* out = (float*)d_out;

    char* ws = (char*)d_ws;
    _Float16* Tf = (_Float16*)(ws);
    _Float16* Ti = (_Float16*)(ws + (1u << 20));
    _Float16* X1 = (_Float16*)(ws + (2u << 20));
    _Float16* X2 = (_Float16*)(ws + (4u << 20));

    // Twiddle tables
    k_twiddles<<<(MF * NL) / 256, 256, 0, stream>>>(Tf, Ti);

    // Stage 1: X1[b] = Tf(128x4096) * f16(x[b])(4096x256)   [x read once]
    k_dft<32><<<NB * (ND / 32), 256, 0, stream>>>(Tf, x, X1, NL, ND);

    // Stage 2: per-mode complex mix into X2
    k_mix<<<NMODE * 16, 128, 0, stream>>>(X1, wr, wi, X2);

    // Stage 3: out[b] = (2/N) * Ti(4096x128) * X2[b](128x256)  [TDM staged]
    k_idft<<<NB * (NL / 128) * (ND / 64), 256, 0, stream>>>(
        Ti, X2, out, 2.0f / (float)NL);
}
